// DA3CrossFrameRKDDistanceLoss_36524401885583
// MI455X (gfx1250) — compile-verified, matrix operand order checked by reading, not run on hardware
//
#include <hip/hip_runtime.h>
#include <hip/hip_bf16.h>

// ---------------------------------------------------------------------------
// Problem constants (from reference)
// ---------------------------------------------------------------------------
#define BB 4          // batch
#define ST 8          // teacher frames
#define SS 4          // student frames
#define PP 1024       // patches
#define DD 1024       // feature dim
#define NREF 256      // ref_perm size (== N)
#define MEXT 4096     // 4 extra frames * P
#define TOPK 4

typedef __attribute__((ext_vector_type(16))) __bf16 v16bf;
typedef __attribute__((ext_vector_type(8)))  float  v8f;

// ---------------------------------------------------------------------------
// helpers
// ---------------------------------------------------------------------------
static __device__ __forceinline__ unsigned short f32_to_bf16(float f) {
    unsigned int u = __float_as_uint(f);
    u += 0x7FFFu + ((u >> 16) & 1u);   // round-to-nearest-even
    return (unsigned short)(u >> 16);
}

static __device__ __forceinline__ float wave_sum32(float v) {
    #pragma unroll
    for (int m = 16; m > 0; m >>= 1) v += __shfl_xor(v, m, 32);
    return v;
}
static __device__ __forceinline__ float wave_max32(float v) {
    #pragma unroll
    for (int m = 16; m > 0; m >>= 1) v = fmaxf(v, __shfl_xor(v, m, 32));
    return v;
}

// ---------------------------------------------------------------------------
// Kernel 1a: gather ref rows (teacher frame 0, via ref_perm), l2-normalize,
//            store bf16.  grid = B*NREF blocks, 256 threads.
// ---------------------------------------------------------------------------
__global__ void norm_ref_kernel(const float* __restrict__ teacher,
                                const int* __restrict__ ref_perm,
                                unsigned short* __restrict__ Abf) {
    const int bid = blockIdx.x;
    const int b = bid / NREF, n = bid % NREF;
    const int tid = threadIdx.x, w = tid >> 5, lane = tid & 31;
    const float* src = teacher + (((size_t)b * ST + 0) * PP + ref_perm[n]) * DD;
    unsigned short* dst = Abf + ((size_t)b * NREF + n) * DD;

    float ssq = 0.f;
    #pragma unroll
    for (int i = 0; i < 4; ++i) { float v = src[tid + i * 256]; ssq += v * v; }
    ssq = wave_sum32(ssq);
    __shared__ float red[8];
    if (lane == 0) red[w] = ssq;
    __syncthreads();
    float tot = red[0] + red[1] + red[2] + red[3] + red[4] + red[5] + red[6] + red[7];
    float scale = 1.0f / fmaxf(sqrtf(tot), 1e-12f);
    #pragma unroll
    for (int i = 0; i < 4; ++i)
        dst[tid + i * 256] = f32_to_bf16(src[tid + i * 256] * scale);
}

// ---------------------------------------------------------------------------
// Kernel 1b: extra_t rows (teacher frames 1,3,5,7), l2-normalize, store bf16.
//            grid = B*MEXT blocks, 256 threads.
// ---------------------------------------------------------------------------
__global__ void norm_extra_kernel(const float* __restrict__ teacher,
                                  unsigned short* __restrict__ Bbf) {
    const int bid = blockIdx.x;
    const int b = bid / MEXT, m = bid % MEXT;
    const int frame = 2 * (m >> 10) + 1;      // EXTRA_FRAMES = {1,3,5,7}
    const int p = m & (PP - 1);
    const int tid = threadIdx.x, w = tid >> 5, lane = tid & 31;
    const float* src = teacher + (((size_t)b * ST + frame) * PP + p) * DD;
    unsigned short* dst = Bbf + ((size_t)b * MEXT + m) * DD;

    float ssq = 0.f;
    #pragma unroll
    for (int i = 0; i < 4; ++i) { float v = src[tid + i * 256]; ssq += v * v; }
    ssq = wave_sum32(ssq);
    __shared__ float red[8];
    if (lane == 0) red[w] = ssq;
    __syncthreads();
    float tot = red[0] + red[1] + red[2] + red[3] + red[4] + red[5] + red[6] + red[7];
    float scale = 1.0f / fmaxf(sqrtf(tot), 1e-12f);
    #pragma unroll
    for (int i = 0; i < 4; ++i)
        dst[tid + i * 256] = f32_to_bf16(src[tid + i * 256] * scale);
}

// ---------------------------------------------------------------------------
// Kernel 2: WMMA GEMM  sim[b] = A(256x1024, bf16) x B^T(4096x1024, bf16)
//   block = 256 threads (8 waves); block tile 32(M) x 64(N); each wave owns
//   one 16x16 tile and runs K=1024 in steps of 32 with v_wmma_f32_16x16x32_bf16.
//   Fragment layout per CDNA5 ISA 7.12.2 (16-bit A 16x32): lane l holds
//   row/col (l&15); element chunks at K = (l>>4)*8 and K = 16+(l>>4)*8.
//   grid = (MEXT/64, NREF/32, B)
// ---------------------------------------------------------------------------
union Frag { v16bf v; uint4 u[2]; };

__global__ void sim_gemm_kernel(const unsigned short* __restrict__ Abf,
                                const unsigned short* __restrict__ Bbf,
                                float* __restrict__ sim) {
    const int b    = blockIdx.z;
    const int tid  = threadIdx.x;
    const int w    = tid >> 5;
    const int lane = tid & 31;
    const int half = lane >> 4;
    const int l16  = lane & 15;

    const int Mbase = blockIdx.y * 32 + (w & 1) * 16;
    const int Nbase = blockIdx.x * 64 + (w >> 1) * 16;

    const unsigned short* Arow = Abf + ((size_t)b * NREF + (Mbase + l16)) * DD;
    const unsigned short* Brow = Bbf + ((size_t)b * MEXT + (Nbase + l16)) * DD;

    v8f c = {};
    #pragma unroll 4
    for (int kk = 0; kk < DD; kk += 32) {
        Frag fa, fb;
        fa.u[0] = *(const uint4*)(Arow + kk +      half * 8);
        fa.u[1] = *(const uint4*)(Arow + kk + 16 + half * 8);
        fb.u[0] = *(const uint4*)(Brow + kk +      half * 8);
        fb.u[1] = *(const uint4*)(Brow + kk + 16 + half * 8);
        c = __builtin_amdgcn_wmma_f32_16x16x32_bf16(
                /*neg_a=*/false, fa.v, /*neg_b=*/false, fb.v,
                /*c_mod=*/(short)0, c, /*reuse_a=*/false, /*reuse_b=*/false);
    }

    // C layout: element e, lanes 0-15 -> M=e, lanes 16-31 -> M=8+e; N = l16.
    float* out = sim + ((size_t)b * NREF + (Mbase + half * 8)) * MEXT + Nbase + l16;
    #pragma unroll
    for (int e = 0; e < 8; ++e) out[(size_t)e * MEXT] = c[e];
}

// ---------------------------------------------------------------------------
// Kernel 3: top-4 along m=4096 per (b,n).  grid = B*NREF blocks, 256 threads.
// ---------------------------------------------------------------------------
__global__ void topk_kernel(const float* __restrict__ sim,
                            int* __restrict__ topk) {
    const int bid = blockIdx.x;                 // b*NREF + n
    const float* row = sim + (size_t)bid * MEXT;
    const int tid = threadIdx.x;

    const float NEG = -3.402823466e38f;
    float v0 = NEG, v1 = NEG, v2 = NEG, v3 = NEG;
    int   i0 = 0,   i1 = 0,   i2 = 0,   i3 = 0;
    for (int m = tid; m < MEXT; m += 256) {
        float v = row[m];
        if (v > v0)      { v3=v2;i3=i2; v2=v1;i2=i1; v1=v0;i1=i0; v0=v;i0=m; }
        else if (v > v1) { v3=v2;i3=i2; v2=v1;i2=i1; v1=v;i1=m; }
        else if (v > v2) { v3=v2;i3=i2; v2=v;i2=m; }
        else if (v > v3) { v3=v;i3=m; }
    }
    __shared__ float sv[1024];
    __shared__ int   si[1024];
    sv[tid*4+0]=v0; si[tid*4+0]=i0; sv[tid*4+1]=v1; si[tid*4+1]=i1;
    sv[tid*4+2]=v2; si[tid*4+2]=i2; sv[tid*4+3]=v3; si[tid*4+3]=i3;
    __syncthreads();
    if (tid == 0) {
        float bv0=NEG,bv1=NEG,bv2=NEG,bv3=NEG; int bi0=0,bi1=0,bi2=0,bi3=0;
        for (int c = 0; c < 1024; ++c) {
            float v = sv[c]; int i = si[c];
            bool g0 = (v > bv0) || (v == bv0 && i < bi0);
            bool g1 = (v > bv1) || (v == bv1 && i < bi1);
            bool g2 = (v > bv2) || (v == bv2 && i < bi2);
            bool g3 = (v > bv3) || (v == bv3 && i < bi3);
            if (g0)      { bv3=bv2;bi3=bi2; bv2=bv1;bi2=bi1; bv1=bv0;bi1=bi0; bv0=v;bi0=i; }
            else if (g1) { bv3=bv2;bi3=bi2; bv2=bv1;bi2=bi1; bv1=v;bi1=i; }
            else if (g2) { bv3=bv2;bi3=bi2; bv2=v;bi2=i; }
            else if (g3) { bv3=v;bi3=i; }
        }
        topk[bid*4+0]=bi0; topk[bid*4+1]=bi1; topk[bid*4+2]=bi2; topk[bid*4+3]=bi3;
    }
}

// ---------------------------------------------------------------------------
// Kernel 4: KL branches.  grid = (NREF, B, 3 shared-teacher), block = 288
//   (9 waves): wave 0 -> d1 row; waves 1-4 -> d2 (j=w-1); waves 5-8 -> d3.
//   One wave32 per KL row of length D=1024 (32 elems/lane in registers).
// ---------------------------------------------------------------------------
__global__ void kl_kernel(const float* __restrict__ teacher,
                          const float* __restrict__ student,
                          const int* __restrict__ ref_perm,
                          const int* __restrict__ shared_perm,
                          const int* __restrict__ topk,
                          float* __restrict__ d1,
                          float* __restrict__ d2,
                          float* __restrict__ d3) {
    const int n = blockIdx.x;
    const int b = blockIdx.y;
    const int z = blockIdx.z;            // 0..2  -> t_idx = 2z+2, s_idx = z+1
    const int t_idx = 2 * (z + 1);
    const int s_idx = z + 1;
    const int w    = threadIdx.x >> 5;   // 0..8
    const int lane = threadIdx.x & 31;

    const int rp = ref_perm[n];
    const int sp = shared_perm[n];
    const float* rt = teacher + (((size_t)b * ST + 0    ) * PP + rp) * DD;
    const float* rs = student + (((size_t)b * SS + 0    ) * PP + rp) * DD;
    const float* st = teacher + (((size_t)b * ST + t_idx) * PP + sp) * DD;
    const float* ss = student + (((size_t)b * SS + s_idx) * PP + sp) * DD;

    const float *xA, *xB, *yA, *yB;
    int j = 0;
    if (w == 0) {
        xA = rt; xB = st; yA = rs; yB = ss;
    } else {
        j = (w <= 4) ? (w - 1) : (w - 5);
        const int m = topk[((size_t)b * NREF + n) * 4 + j];
        const float* sh = teacher +
            (((size_t)b * ST + (2 * (m >> 10) + 1)) * PP + (m & (PP - 1))) * DD;
        if (w <= 4) { xA = rt; xB = sh; yA = rs; yB = sh; }
        else        { xA = st; xB = sh; yA = ss; yB = sh; }
    }

    float xt[32], xs[32];
    #pragma unroll
    for (int i = 0; i < 32; ++i) {
        const int d = i * 32 + lane;
        xt[i] = xA[d] - xB[d];           // TEMP == 1.0
        xs[i] = yA[d] - yB[d];
    }

    float mx = -3.402823466e38f, my = -3.402823466e38f;
    #pragma unroll
    for (int i = 0; i < 32; ++i) { mx = fmaxf(mx, xt[i]); my = fmaxf(my, xs[i]); }
    mx = wave_max32(mx);
    my = wave_max32(my);

    float set = 0.f, ses = 0.f;
    #pragma unroll
    for (int i = 0; i < 32; ++i) { set += __expf(xt[i] - mx); ses += __expf(xs[i] - my); }
    set = wave_sum32(set);
    ses = wave_sum32(ses);
    const float lt = __logf(set), ls = __logf(ses);

    float acc = 0.f;
    #pragma unroll
    for (int i = 0; i < 32; ++i) {
        const float a = xt[i] - mx, q = xs[i] - my;
        acc += __expf(a) * ((a - lt) - (q - ls));
    }
    acc = wave_sum32(acc);
    const float kl = acc / set;

    const float ax = fabsf(kl);          // smooth_l1_zero, beta = 0.5
    const float h  = (ax < 0.5f) ? (ax * ax) : (ax - 0.25f);

    if (lane == 0) {
        const size_t base = ((size_t)z * BB + b) * NREF + n;
        if (w == 0)      d1[base] = h;
        else if (w <= 4) d2[base * 4 + j] = h;
        else             d3[base * 4 + j] = h;
    }
}

// ---------------------------------------------------------------------------
// Kernel 5: deterministic final reduction -> scalar loss.
// ---------------------------------------------------------------------------
__global__ void finalize_kernel(const float* __restrict__ d1,
                                const float* __restrict__ d2,
                                const float* __restrict__ d3,
                                float* __restrict__ out) {
    const int tid = threadIdx.x, w = tid >> 5, lane = tid & 31;
    const int N1 = 3 * BB * NREF;            // 3072
    const int N2 = 3 * BB * NREF * TOPK;     // 12288
    float s1 = 0.f, s2 = 0.f, s3 = 0.f;
    for (int i = tid; i < N1; i += 256) s1 += d1[i];
    for (int i = tid; i < N2; i += 256) s2 += d2[i];
    for (int i = tid; i < N2; i += 256) s3 += d3[i];
    s1 = wave_sum32(s1); s2 = wave_sum32(s2); s3 = wave_sum32(s3);
    __shared__ float r1[8], r2[8], r3[8];
    if (lane == 0) { r1[w] = s1; r2[w] = s2; r3[w] = s3; }
    __syncthreads();
    if (tid == 0) {
        float t1 = 0.f, t2 = 0.f, t3 = 0.f;
        for (int i = 0; i < 8; ++i) { t1 += r1[i]; t2 += r2[i]; t3 += r3[i]; }
        out[0] = t1 / (float)N1 + t2 / (float)N2 + t3 / (float)N2;
    }
}

// ---------------------------------------------------------------------------
// launch
// ---------------------------------------------------------------------------
extern "C" void kernel_launch(void* const* d_in, const int* in_sizes, int n_in,
                              void* d_out, int out_size, void* d_ws, size_t ws_size,
                              hipStream_t stream) {
    (void)in_sizes; (void)n_in; (void)out_size; (void)ws_size;
    const float* teacher     = (const float*)d_in[0];
    const float* student     = (const float*)d_in[1];
    const int*   ref_perm    = (const int*)d_in[2];
    const int*   shared_perm = (const int*)d_in[3];
    float*       out         = (float*)d_out;

    char* ws = (char*)d_ws;
    size_t off = 0;
    auto alloc = [&](size_t bytes) -> void* {
        off = (off + 255) & ~(size_t)255;
        void* p = ws + off;
        off += bytes;
        return p;
    };
    unsigned short* Abf = (unsigned short*)alloc((size_t)BB * NREF * DD * 2);  //  2 MB
    unsigned short* Bbf = (unsigned short*)alloc((size_t)BB * MEXT * DD * 2);  // 32 MB
    float* sim  = (float*)alloc((size_t)BB * NREF * MEXT * 4);                 // 16 MB
    int*   topk = (int*)  alloc((size_t)BB * NREF * TOPK * 4);
    float* d1   = (float*)alloc((size_t)3 * BB * NREF * 4);
    float* d2   = (float*)alloc((size_t)3 * BB * NREF * TOPK * 4);
    float* d3   = (float*)alloc((size_t)3 * BB * NREF * TOPK * 4);

    norm_ref_kernel  <<<BB * NREF, 256, 0, stream>>>(teacher, ref_perm, Abf);
    norm_extra_kernel<<<BB * MEXT, 256, 0, stream>>>(teacher, Bbf);
    sim_gemm_kernel  <<<dim3(MEXT / 64, NREF / 32, BB), 256, 0, stream>>>(Abf, Bbf, sim);
    topk_kernel      <<<BB * NREF, 256, 0, stream>>>(sim, topk);
    kl_kernel        <<<dim3(NREF, BB, 3), 288, 0, stream>>>(
        teacher, student, ref_perm, shared_perm, topk, d1, d2, d3);
    finalize_kernel  <<<1, 256, 0, stream>>>(d1, d2, d3, out);
}
